// GRU_classifier_49254684950849
// MI455X (gfx1250) — compile-verified, matrix operand order checked by reading, not run on hardware
//
#include <hip/hip_runtime.h>

#define BATCH       4096
#define INPUT_SIZE  1024
#define H           2048
#define OUTS        2

// ---- WMMA vector types (gfx1250, wave32) ----
typedef __attribute__((ext_vector_type(16))) __bf16 v16bf;
typedef __attribute__((ext_vector_type(8)))  float  v8f;

// GEMM tiling: block tile 128(M) x 64(N), K-step 32, 256 threads = 8 waves.
// Wave grid 4(M) x 2(N): each wave owns a 32x32 patch = 2x2 WMMA 16x16 tiles.
#define BM 128
#define BN 64
#define BK 32
#define AS_STRIDE 40   // bf16 elements per LDS row (32 + 8 pad -> 80B rows, 16B aligned)
#define BS_STRIDE 40

__device__ __forceinline__ unsigned pack_bf16x2(float lo, float hi) {
    unsigned a = __builtin_bit_cast(unsigned, lo);
    unsigned b = __builtin_bit_cast(unsigned, hi);
    return (a >> 16) | (b & 0xFFFF0000u);
}

__device__ __forceinline__ unsigned short f32_to_bf16_bits(float f) {
    return (unsigned short)(__builtin_bit_cast(unsigned, f) >> 16);
}

__device__ __forceinline__ float sigmoidf_fast(float v) {
    return 1.0f / (1.0f + __expf(-v));
}

union Frag {
    v16bf v;
    uint2 u2[4];
};

// Load a 16-lane-row x 32(K) bf16 fragment from LDS in the ISA A/B layout:
// lanes 0-15 hold K = {0..7, 16..23}, lanes 16-31 hold K = {8..15, 24..31}.
__device__ __forceinline__ v16bf load_frag(const unsigned short* lds, int row,
                                           int stride, int khalf) {
    Frag f;
    const int base = row * stride + khalf * 8;
    f.u2[0] = *(const uint2*)&lds[base];
    f.u2[1] = *(const uint2*)&lds[base + 4];
    f.u2[2] = *(const uint2*)&lds[base + 16];
    f.u2[3] = *(const uint2*)&lds[base + 20];
    return f.v;
}

// ---------------------------------------------------------------------------
// Stage 1: hz = Wh_z @ h, hr = Wh_r @ h   (tiny matvecs, one wave per row)
// ---------------------------------------------------------------------------
__global__ __launch_bounds__(256) void matvec_hz_hr(
    const float* __restrict__ Whz, const float* __restrict__ Whr,
    const float* __restrict__ h, float* __restrict__ hz, float* __restrict__ hr) {
    const int lane = threadIdx.x & 31;
    const int row  = blockIdx.x * 8 + (threadIdx.x >> 5);
    const float* rz = Whz + (size_t)row * H;
    const float* rr = Whr + (size_t)row * H;
    float s0 = 0.f, s1 = 0.f;
    for (int k = lane; k < H; k += 32) {
        const float hv = h[k];
        s0 += rz[k] * hv;
        s1 += rr[k] * hv;
    }
    #pragma unroll
    for (int off = 16; off; off >>= 1) {
        s0 += __shfl_down(s0, off, 32);
        s1 += __shfl_down(s1, off, 32);
    }
    if (lane == 0) { hz[row] = s0; hr[row] = s1; }
}

// ---------------------------------------------------------------------------
// Stage 2/3: WMMA GEMM  C[B, H] = epilogue( A @ W^T ),  W is [H, K] row-major.
//   MODE 0 (Z): zt = sigmoid(acc + bias[n] + extra[n])            -> f32 out
//   MODE 1 (R): g  = bf16( sigmoid(acc + bias[n]+extra[n])*h[n] ) -> bf16 out
//   MODE 2 (H): ah = acc + bias[n]                                -> f32 out
//   MODE 3 (U): h_new = zt*h[n] + (1-zt)*tanh(ah + acc)           -> f32 out
// MODE 3's A matrix is already bf16 -> staged with ASYNC global->LDS copies
// (no VGPR round-trip, tracked by ASYNCcnt). Modes 0-2 convert f32 x on the
// fly through VGPRs.
// ---------------------------------------------------------------------------
template <int MODE>
__global__ __launch_bounds__(256) void gemm_kernel(
    const float* __restrict__ Af32,
    const unsigned short* __restrict__ Abf16,
    const float* __restrict__ W,
    const float* __restrict__ bias,
    const float* __restrict__ extra,
    const float* __restrict__ hvec,
    const float* __restrict__ zt,
    const float* __restrict__ ah,
    float* __restrict__ out_f32,
    unsigned short* __restrict__ out_bf16,
    int K) {

    __shared__ unsigned short As[BM * AS_STRIDE];
    __shared__ unsigned short Bs[BN * BS_STRIDE];

    const int tid   = threadIdx.x;
    const int lane  = tid & 31;
    const int wave  = tid >> 5;
    const int waveM = (wave & 3) * 32;
    const int waveN = (wave >> 2) * 32;
    const int tileM = blockIdx.y * BM;
    const int tileN = blockIdx.x * BN;

    v8f acc[2][2] = {{{}, {}}, {{}, {}}};

    const int numK = K / BK;
    for (int kt = 0; kt < numK; ++kt) {
        const int k0 = kt * BK;

        // --- stage A tile (BM x BK) into LDS as bf16 ---
        if constexpr (MODE == 3) {
            // A is bf16 in memory: async DMA straight into LDS, 16B chunks.
            // Row = 64B of payload = 4 chunks; 512 chunks over 256 threads.
            #pragma unroll
            for (int i = 0; i < 2; ++i) {
                const int idx = tid + i * 256;
                const int row = idx >> 2;
                const int c   = idx & 3;
                const unsigned long long gaddr =
                    (unsigned long long)(Abf16 + (size_t)(tileM + row) * K + k0) +
                    (unsigned long long)(c * 16);
                const unsigned ldsaddr =
                    (unsigned)(unsigned long long)(&As[row * AS_STRIDE]) + c * 16;
                asm volatile("global_load_async_to_lds_b128 %0, %1, off"
                             :: "v"(ldsaddr), "v"(gaddr) : "memory");
                __builtin_prefetch(Abf16 + (size_t)(tileM + row) * K + k0 + BK, 0, 1);
            }
        } else {
            #pragma unroll
            for (int i = 0; i < (BM * BK / 4) / 256; ++i) {
                const int idx = tid + i * 256;
                const int row = idx >> 3;          // BK/4 == 8 quads per row
                const int c4  = (idx & 7) << 2;
                const float4 f = *(const float4*)(Af32 + (size_t)(tileM + row) * K + k0 + c4);
                unsigned short* dst = &As[row * AS_STRIDE + c4];
                *(unsigned*)(dst)     = pack_bf16x2(f.x, f.y);
                *(unsigned*)(dst + 2) = pack_bf16x2(f.z, f.w);
                __builtin_prefetch(Af32 + (size_t)(tileM + row) * K + k0 + BK + c4, 0, 1);
            }
        }
        // --- stage W tile (BN rows of W = output columns, BK wide) ---
        #pragma unroll
        for (int i = 0; i < (BN * BK / 4) / 256; ++i) {
            const int idx = tid + i * 256;
            const int row = idx >> 3;
            const int c4  = (idx & 7) << 2;
            const float4 f = *(const float4*)(W + (size_t)(tileN + row) * K + k0 + c4);
            unsigned short* dst = &Bs[row * BS_STRIDE + c4];
            *(unsigned*)(dst)     = pack_bf16x2(f.x, f.y);
            *(unsigned*)(dst + 2) = pack_bf16x2(f.z, f.w);
            __builtin_prefetch(W + (size_t)(tileN + row) * K + k0 + BK + c4, 0, 1);
        }
        if constexpr (MODE == 3) {
            asm volatile("s_wait_asynccnt 0" ::: "memory");
        }
        __syncthreads();

        // --- 2x2 WMMA tiles per wave ---
        const int frow  = lane & 15;
        const int khalf = lane >> 4;
        v16bf afrag[2], bfrag[2];
        #pragma unroll
        for (int t = 0; t < 2; ++t) {
            afrag[t] = load_frag(As, waveM + t * 16 + frow, AS_STRIDE, khalf);
            bfrag[t] = load_frag(Bs, waveN + t * 16 + frow, BS_STRIDE, khalf);
        }
        #pragma unroll
        for (int tm = 0; tm < 2; ++tm)
            #pragma unroll
            for (int tn = 0; tn < 2; ++tn)
                acc[tm][tn] = __builtin_amdgcn_wmma_f32_16x16x32_bf16(
                    false, afrag[tm], false, bfrag[tn],
                    (short)0, acc[tm][tn], false, false);
        __syncthreads();
    }

    // --- epilogue: C layout = VGPR r -> M = r + 8*(lane>=16), N = lane&15 ---
    const int mOff = ((lane >> 4) << 3);
    const int nOff = (lane & 15);
    #pragma unroll
    for (int tm = 0; tm < 2; ++tm) {
        #pragma unroll
        for (int tn = 0; tn < 2; ++tn) {
            #pragma unroll
            for (int r = 0; r < 8; ++r) {
                const int m = tileM + waveM + tm * 16 + r + mOff;
                const int n = tileN + waveN + tn * 16 + nOff;
                const size_t idx = (size_t)m * H + n;
                const float v = acc[tm][tn][r];
                if constexpr (MODE == 0) {
                    out_f32[idx] = sigmoidf_fast(v + bias[n] + extra[n]);
                } else if constexpr (MODE == 1) {
                    const float rt = sigmoidf_fast(v + bias[n] + extra[n]);
                    out_bf16[idx] = f32_to_bf16_bits(rt * hvec[n]);
                } else if constexpr (MODE == 2) {
                    out_f32[idx] = v + bias[n];
                } else {
                    const float t = ah[idx] + v;
                    const float z = zt[idx];
                    out_f32[idx] = z * hvec[n] + (1.0f - z) * tanhf(t);
                }
            }
        }
    }
}

// ---------------------------------------------------------------------------
// Stage 4: y[b, o] = sum_k h_new[b,k] * Wy[o,k] + by[o]   (OUT = 2)
// ---------------------------------------------------------------------------
__global__ __launch_bounds__(256) void head_kernel(
    const float* __restrict__ hn, const float* __restrict__ Wy,
    const float* __restrict__ by, float* __restrict__ y) {
    const int lane = threadIdx.x & 31;
    const int b    = blockIdx.x * 8 + (threadIdx.x >> 5);
    const float* row = hn + (size_t)b * H;
    float s0 = 0.f, s1 = 0.f;
    for (int k = lane; k < H; k += 32) {
        const float v = row[k];
        s0 += v * Wy[k];
        s1 += v * Wy[H + k];
    }
    #pragma unroll
    for (int off = 16; off; off >>= 1) {
        s0 += __shfl_down(s0, off, 32);
        s1 += __shfl_down(s1, off, 32);
    }
    if (lane == 0) {
        y[(size_t)b * OUTS + 0] = s0 + by[0];
        y[(size_t)b * OUTS + 1] = s1 + by[1];
    }
}

// ---------------------------------------------------------------------------
extern "C" void kernel_launch(void* const* d_in, const int* in_sizes, int n_in,
                              void* d_out, int out_size, void* d_ws, size_t ws_size,
                              hipStream_t stream) {
    const float* x    = (const float*)d_in[0];
    const float* h    = (const float*)d_in[1];
    const float* Wx_z = (const float*)d_in[2];
    const float* bx_z = (const float*)d_in[3];
    const float* Wx_r = (const float*)d_in[4];
    const float* bx_r = (const float*)d_in[5];
    const float* Wx_h = (const float*)d_in[6];
    const float* bx_h = (const float*)d_in[7];
    const float* Wh_z = (const float*)d_in[8];
    const float* Wh_r = (const float*)d_in[9];
    const float* Wh_h = (const float*)d_in[10];
    const float* Wy   = (const float*)d_in[11];
    const float* by   = (const float*)d_in[12];
    float* y = (float*)d_out;

    // workspace layout
    float* hz = (float*)d_ws;                 // [H]
    float* hr = hz + H;                       // [H]
    float* zt = hr + H;                       // [B,H]
    float* ah = zt + (size_t)BATCH * H;       // [B,H]
    float* hn = ah + (size_t)BATCH * H;       // [B,H]
    unsigned short* g = (unsigned short*)(hn + (size_t)BATCH * H);  // [B,H] bf16

    const dim3 blk(256);
    const dim3 gemm_grid(H / BN, BATCH / BM);

    // stage 1: hidden-state matvecs (h is unbatched)
    matvec_hz_hr<<<H / 8, blk, 0, stream>>>(Wh_z, Wh_r, h, hz, hr);

    // stage 2: the three input-gate GEMMs (K = 1024, A = x in f32 -> bf16)
    gemm_kernel<0><<<gemm_grid, blk, 0, stream>>>(x, nullptr, Wx_z, bx_z, hz, h,
                                                  nullptr, nullptr, zt, nullptr,
                                                  INPUT_SIZE);
    gemm_kernel<1><<<gemm_grid, blk, 0, stream>>>(x, nullptr, Wx_r, bx_r, hr, h,
                                                  nullptr, nullptr, nullptr, g,
                                                  INPUT_SIZE);
    gemm_kernel<2><<<gemm_grid, blk, 0, stream>>>(x, nullptr, Wx_h, bx_h, nullptr, h,
                                                  nullptr, nullptr, ah, nullptr,
                                                  INPUT_SIZE);

    // stage 3: candidate GEMM + GRU update (K = 2048, A = g in bf16, async->LDS)
    gemm_kernel<3><<<gemm_grid, blk, 0, stream>>>(nullptr, g, Wh_h, nullptr, nullptr, h,
                                                  zt, ah, hn, nullptr,
                                                  H);

    // stage 4: output head
    head_kernel<<<BATCH / 8, blk, 0, stream>>>(hn, Wy, by, y);
}
